// VQVAE_1563368096098
// MI455X (gfx1250) — compile-verified, matrix operand order checked by reading, not run on hardware
//
#include <hip/hip_runtime.h>
#include <hip/hip_bf16.h>

typedef __attribute__((ext_vector_type(16))) __bf16 v16bf;
typedef __attribute__((ext_vector_type(8)))  float  v8f;

union Frag16 { uint4 q[2]; v16bf v; unsigned short u[16]; };

__device__ __forceinline__ unsigned short f2bf(float f) {
  unsigned int u = __float_as_uint(f);
  unsigned int r = u + 0x7FFFu + ((u >> 16) & 1u);
  return (unsigned short)(r >> 16);
}
__device__ __forceinline__ float bf2f(unsigned short h) {
  return __uint_as_float(((unsigned int)h) << 16);
}

// ---------------------------------------------------------------- zero fill
__global__ void k_zero(uint4* __restrict__ p, long long n16) {
  long long i = (long long)blockIdx.x * blockDim.x + threadIdx.x;
  long long stride = (long long)gridDim.x * blockDim.x;
  uint4 z = make_uint4(0u, 0u, 0u, 0u);
  for (; i < n16; i += stride) p[i] = z;
}

// ------------------------------------------------- weight packing (encoder)
// src OIHW fp32 [OC][IC][KH][KW] -> dst bf16 [OC][ (ky*KW+kx)*IC + ic ]
__global__ void k_pack_oihw(const float* __restrict__ src, unsigned short* __restrict__ dst,
                            int OC, int IC, int KH, int KW) {
  int K = IC * KH * KW;
  int total = OC * K;
  for (int i = blockIdx.x * blockDim.x + threadIdx.x; i < total; i += gridDim.x * blockDim.x) {
    int oc = i / K, k = i % K;
    int tap = k / IC, ic = k % IC;
    int ky = tap / KW, kx = tap % KW;
    dst[i] = f2bf(src[((oc * IC + ic) * KH + ky) * KW + kx]);
  }
}

// ------------------------------------------------- weight packing (decoder)
// src (torch ConvTranspose) [IC][OC][KH][KW] fp32, flip kh/kw + swap I/O:
// dst bf16 [OC][ (ky*KW+kx)*IC + ic ]
__global__ void k_pack_convT(const float* __restrict__ src, unsigned short* __restrict__ dst,
                             int IC, int OC, int KH, int KW) {
  int K = IC * KH * KW;
  int total = OC * K;
  for (int i = blockIdx.x * blockDim.x + threadIdx.x; i < total; i += gridDim.x * blockDim.x) {
    int oc = i / K, k = i % K;
    int tap = k / IC, ic = k % IC;
    int ky = tap / KW, kx = tap % KW;
    dst[i] = f2bf(src[((ic * OC + oc) * KH + (KH - 1 - ky)) * KW + (KW - 1 - kx)]);
  }
}

// ------------------------------------------ codebook transpose + 0.5*||e||^2
__global__ void k_pack_emb(const float* __restrict__ emb, unsigned short* __restrict__ embT,
                           float* __restrict__ embN, int D, int Kn) {
  int n = blockIdx.x * blockDim.x + threadIdx.x;
  if (n >= Kn) return;
  float s = 0.f;
  for (int d = 0; d < D; d++) {
    float v = emb[d * Kn + n];
    embT[n * D + d] = f2bf(v);
    s += v * v;
  }
  embN[n] = 0.5f * s;
}

// ---------------------------------------------------- encoder conv1 (Cin=1)
// x [32][256][256] fp32 -> A1 NHWC padded bf16 [32][130][130][128], 4x4 s2 p1 + ReLU
__global__ void k_conv1(const float* __restrict__ x, const float* __restrict__ w,
                        const float* __restrict__ b, unsigned short* __restrict__ out) {
  __shared__ float sw[128 * 16];
  __shared__ float sb[128];
  for (int i = threadIdx.x; i < 2048; i += blockDim.x) sw[i] = w[i];
  for (int i = threadIdx.x; i < 128; i += blockDim.x) sb[i] = b[i];
  __syncthreads();
  int gid = blockIdx.x * blockDim.x + threadIdx.x;   // 32*128*128
  if (gid >= 32 * 128 * 128) return;
  int ox = gid & 127, oy = (gid >> 7) & 127, n = gid >> 14;
  float t[16];
  #pragma unroll
  for (int ky = 0; ky < 4; ky++)
    #pragma unroll
    for (int kx = 0; kx < 4; kx++) {
      int iy = 2 * oy - 1 + ky, ix = 2 * ox - 1 + kx;
      float v = 0.f;
      if (iy >= 0 && iy < 256 && ix >= 0 && ix < 256)
        v = x[(long long)n * 65536 + iy * 256 + ix];
      t[ky * 4 + kx] = v;
    }
  unsigned short* op = out + (((long long)n * 130 + (oy + 1)) * 130 + (ox + 1)) * 128;
  for (int oc0 = 0; oc0 < 128; oc0 += 8) {
    uint4 pk;
    unsigned short* pu = (unsigned short*)&pk;
    #pragma unroll
    for (int j = 0; j < 8; j++) {
      int oc = oc0 + j;
      float acc = sb[oc];
      const float* wr = &sw[oc * 16];
      #pragma unroll
      for (int e = 0; e < 16; e++) acc = fmaf(t[e], wr[e], acc);
      pu[j] = f2bf(fmaxf(acc, 0.f));
    }
    *(uint4*)(op + oc0) = pk;
  }
}

// ------------------------------------------- generic implicit-GEMM conv (WMMA)
// src: NHWC padded bf16 [N][Hp][Wp][CIN] (pad baked in, conv stride SS)
// wgt: bf16 [OC][K], K = KH*KW*CIN ordered (tap-major, ic-minor)
// One wave = 16 pixels x 64 output channels (4 WMMA accumulators sharing one
// A fragment), software-pipelined one K-chunk ahead. Block (32,4): the 4 waves
// cover 4 consecutive M-tiles. gridDim.y covers OC in groups of 64.
template <int KH, int KW, int SS, int CIN>
__global__ void k_conv_wmma(const unsigned short* __restrict__ src, int Hp, int Wp,
                            const unsigned short* __restrict__ wgt,
                            const float* __restrict__ bias, int relu,
                            int Ho, int Wo,
                            unsigned short* __restrict__ out, int outHp, int outWp,
                            int po, int so, int Cout) {
  constexpr int K = KH * KW * CIN;
  const int lane = threadIdx.x;
  const int half = lane >> 4;
  const int m = lane & 15;

  const int wT = Wo >> 4;
  int gmt = blockIdx.x * 4 + threadIdx.y;            // global M-tile
  const int oxT = gmt % wT;  gmt /= wT;
  const int oy  = gmt % Ho;
  const int n   = gmt / Ho;
  const int ocBase = blockIdx.y * 64;

  const int px = oxT * 16 + m;  // A-matrix row pixel held by this lane
  const unsigned short* ab =
      src + (((long long)n * Hp + (long long)oy * SS) * Wp + (long long)px * SS) * CIN;
  const unsigned short* wrow[4];
  #pragma unroll
  for (int j = 0; j < 4; j++) wrow[j] = wgt + (long long)(ocBase + j * 16 + m) * K;

  // fragment loaders: K element set for this lane = kc + r*16 + half*8 + {0..7}
  auto loadA = [&](Frag16& a, int kc) {
    #pragma unroll
    for (int r = 0; r < 2; r++) {
      const int kidx = kc + r * 16 + half * 8;
      const int tap = kidx / CIN;                    // CIN is a power of two
      const int ic  = kidx % CIN;
      const int ky = tap / KW, kx = tap % KW;
      a.q[r] = *(const uint4*)(ab + (ky * Wp + kx) * CIN + ic);
    }
  };
  auto loadB = [&](Frag16* b, int kc) {
    #pragma unroll
    for (int j = 0; j < 4; j++)
      #pragma unroll
      for (int r = 0; r < 2; r++)
        b[j].q[r] = *(const uint4*)(wrow[j] + kc + r * 16 + half * 8);
  };

  v8f c[4] = {{}, {}, {}, {}};
  Frag16 a, b[4];
  loadA(a, 0);
  loadB(b, 0);
  for (int kc = 0; kc < K - 32; kc += 32) {
    Frag16 an, bn[4];
    loadA(an, kc + 32);                              // issue next chunk first
    loadB(bn, kc + 32);
    #pragma unroll
    for (int j = 0; j < 4; j++)
      c[j] = __builtin_amdgcn_wmma_f32_16x16x32_bf16(false, a.v, false, b[j].v,
                                                     (short)0, c[j], false, false);
    a = an;
    #pragma unroll
    for (int j = 0; j < 4; j++) b[j] = bn[j];
  }
  #pragma unroll
  for (int j = 0; j < 4; j++)
    c[j] = __builtin_amdgcn_wmma_f32_16x16x32_bf16(false, a.v, false, b[j].v,
                                                   (short)0, c[j], false, false);

  #pragma unroll
  for (int j = 0; j < 4; j++) {
    const int oc = ocBase + j * 16 + m;              // C column held by this lane
    const float bv = bias[oc];
    #pragma unroll
    for (int r = 0; r < 8; r++) {
      float v = c[j][r] + bv;
      if (relu) v = fmaxf(v, 0.f);
      const int prow = oxT * 16 + r + 8 * half;      // C-matrix row -> pixel
      const long long oaddr =
          (((long long)n * outHp + (po + (long long)oy * so)) * outWp +
           (po + (long long)prow * so)) * Cout + oc;
      out[oaddr] = f2bf(v);
    }
  }
}

// ---------------------------------------------- VQ: scores GEMM + argmax (WMMA)
// z [131072][64] bf16, embT [512][64] bf16, embN [512] fp32.
// score = z.e - 0.5||e||^2 ; argmax(score) == argmin(distance).
__global__ void k_vq_argmin(const unsigned short* __restrict__ z,
                            const unsigned short* __restrict__ embT,
                            const float* __restrict__ embN,
                            int* __restrict__ idxOut) {
  const int lane = threadIdx.x;
  const int half = lane >> 4;
  const int m = lane & 15;
  const long long rowBase = (long long)blockIdx.x * 16;

  Frag16 a0, a1;
  const unsigned short* zr = z + (rowBase + m) * 64;
  #pragma unroll
  for (int r = 0; r < 2; r++) {
    a0.q[r] = *(const uint4*)(zr + 0  + r * 16 + half * 8);
    a1.q[r] = *(const uint4*)(zr + 32 + r * 16 + half * 8);
  }

  auto loadB = [&](Frag16& b0, Frag16& b1, int t) {
    const unsigned short* er = embT + (t * 16 + m) * 64;
    #pragma unroll
    for (int r = 0; r < 2; r++) {
      b0.q[r] = *(const uint4*)(er + 0  + r * 16 + half * 8);
      b1.q[r] = *(const uint4*)(er + 32 + r * 16 + half * 8);
    }
  };

  float best[8];
  int   bidx[8];
  #pragma unroll
  for (int r = 0; r < 8; r++) { best[r] = -3.4e38f; bidx[r] = 0; }

  Frag16 b0, b1;
  loadB(b0, b1, 0);
  for (int t = 0; t < 32; t++) {
    Frag16 nb0, nb1;
    if (t < 31) loadB(nb0, nb1, t + 1);              // pipeline next tile
    v8f c = {};
    c = __builtin_amdgcn_wmma_f32_16x16x32_bf16(false, a0.v, false, b0.v, (short)0, c, false, false);
    c = __builtin_amdgcn_wmma_f32_16x16x32_bf16(false, a1.v, false, b1.v, (short)0, c, false, false);
    const int col = t * 16 + m;
    const float nrm = embN[col];
    #pragma unroll
    for (int r = 0; r < 8; r++) {
      float s = c[r] - nrm;
      if (s > best[r]) { best[r] = s; bidx[r] = col; }
    }
    b0 = nb0; b1 = nb1;
  }
  // per-row argmax lives spread across 16 lanes (same half) -> shfl_xor reduce
  #pragma unroll
  for (int off = 1; off < 16; off <<= 1) {
    #pragma unroll
    for (int r = 0; r < 8; r++) {
      float ov = __shfl_xor(best[r], off, 32);
      int   oi = __shfl_xor(bidx[r], off, 32);
      if (ov > best[r] || (ov == best[r] && oi < bidx[r])) { best[r] = ov; bidx[r] = oi; }
    }
  }
  if (m == 0) {
    #pragma unroll
    for (int r = 0; r < 8; r++) idxOut[rowBase + r + 8 * half] = bidx[r];
  }
}

// ---------------- VQ gather: quant -> dilated decoder input, sse + histogram
__global__ void k_vq_gather(const int* __restrict__ idx,
                            const unsigned short* __restrict__ embT,
                            const unsigned short* __restrict__ z,
                            unsigned short* __restrict__ d1in,   // [32][130][130][64]
                            float* __restrict__ counts, float* __restrict__ sse) {
  int r = blockIdx.x * blockDim.x + threadIdx.x;
  if (r >= 131072) return;
  const int id = idx[r];
  const int ix = r & 63, iy = (r >> 6) & 63, n = r >> 12;
  const uint4* q  = (const uint4*)(embT + (long long)id * 64);
  const uint4* zp = (const uint4*)(z + (long long)r * 64);
  uint4* op = (uint4*)(d1in + (((long long)n * 130 + (1 + 2 * iy)) * 130 + (1 + 2 * ix)) * 64);
  float acc = 0.f;
  #pragma unroll
  for (int j = 0; j < 8; j++) {
    uint4 qv = q[j], zv = zp[j];
    op[j] = qv;                                     // quant value == codebook row
    const unsigned short* qu = (const unsigned short*)&qv;
    const unsigned short* zu = (const unsigned short*)&zv;
    #pragma unroll
    for (int e = 0; e < 8; e++) {
      float d = bf2f(qu[e]) - bf2f(zu[e]);
      acc = fmaf(d, d, acc);
    }
  }
  atomicAdd(&counts[id], 1.0f);
  for (int off = 16; off > 0; off >>= 1) acc += __shfl_down(acc, off, 32);
  if ((threadIdx.x & 31) == 0) atomicAdd(sse, acc);
}

// ------------------------------------------------ loss + perplexity finalize
__global__ void k_finalize(const float* __restrict__ counts, const float* __restrict__ sse,
                           float* __restrict__ out_loss, float* __restrict__ out_perp) {
  __shared__ float red[512];
  int t = threadIdx.x;
  float a = counts[t] * (1.0f / 131072.0f);
  red[t] = -a * __logf(a + 1e-10f);
  __syncthreads();
  for (int s = 256; s > 0; s >>= 1) {
    if (t < s) red[t] += red[t + s];
    __syncthreads();
  }
  if (t == 0) {
    *out_perp = __expf(red[0]);
    *out_loss = 1.25f * sse[0] * (1.0f / (131072.0f * 64.0f));  // q_loss + beta*e_loss
  }
}

// --------------------------------------------------- decoder convT3 (Cout=1)
__global__ void k_convT3(const unsigned short* __restrict__ g,   // [32][258][258][64]
                         const unsigned short* __restrict__ w,   // [576] packed (tap,ic)
                         const float* __restrict__ b, float* __restrict__ out) {
  __shared__ float sw[576];
  for (int i = threadIdx.x; i < 576; i += blockDim.x) sw[i] = bf2f(w[i]);
  __syncthreads();
  long long gid = (long long)blockIdx.x * blockDim.x + threadIdx.x;
  if (gid >= 32LL * 256 * 256) return;
  int ox = (int)(gid & 255), oy = (int)((gid >> 8) & 255), n = (int)(gid >> 16);
  float acc = b[0];
  #pragma unroll
  for (int ky = 0; ky < 3; ky++)
    #pragma unroll
    for (int kx = 0; kx < 3; kx++) {
      const uint4* p = (const uint4*)(g + (((long long)n * 258 + (oy + ky)) * 258 + (ox + kx)) * 64);
      const float* wr = &sw[(ky * 3 + kx) * 64];
      #pragma unroll
      for (int j = 0; j < 8; j++) {
        uint4 v = p[j];
        const unsigned short* u = (const unsigned short*)&v;
        #pragma unroll
        for (int e = 0; e < 8; e++) acc = fmaf(bf2f(u[e]), wr[j * 8 + e], acc);
      }
    }
  out[gid] = acc;
}

// ===========================================================================
extern "C" void kernel_launch(void* const* d_in, const int* in_sizes, int n_in,
                              void* d_out, int out_size, void* d_ws, size_t ws_size,
                              hipStream_t stream) {
  const float* x   = (const float*)d_in[0];
  const float* ew1 = (const float*)d_in[1];
  const float* eb1 = (const float*)d_in[2];
  const float* ew2 = (const float*)d_in[3];
  const float* eb2 = (const float*)d_in[4];
  const float* ew3 = (const float*)d_in[5];
  const float* eb3 = (const float*)d_in[6];
  const float* emb = (const float*)d_in[7];
  const float* dw1 = (const float*)d_in[8];
  const float* db1 = (const float*)d_in[9];
  const float* dw2 = (const float*)d_in[10];
  const float* db2 = (const float*)d_in[11];
  const float* dw3 = (const float*)d_in[12];
  const float* db3 = (const float*)d_in[13];
  float* out = (float*)d_out;

  char* ws = (char*)d_ws;
  size_t off = 0;
  auto carve = [&](size_t bytes) -> char* {
    char* p = ws + off;
    off = (off + bytes + 255) & ~(size_t)255;
    return p;
  };
  unsigned short* W2p  = (unsigned short*)carve((size_t)64 * 2048 * 2);        // conv2 packed
  unsigned short* W3p  = (unsigned short*)carve((size_t)64 * 576 * 2);         // conv3 packed
  unsigned short* DW1p = (unsigned short*)carve((size_t)128 * 576 * 2);        // convT1 packed
  unsigned short* DW2p = (unsigned short*)carve((size_t)64 * 1152 * 2);        // convT2 packed
  unsigned short* DW3p = (unsigned short*)carve((size_t)576 * 2);              // convT3 packed
  unsigned short* EmbT = (unsigned short*)carve((size_t)512 * 64 * 2);
  float*          EmbN = (float*)carve((size_t)512 * 4);
  float*          Counts = (float*)carve((size_t)512 * 4 + 16);                // + SSE
  float*          SSE  = Counts + 512;
  int*            Idx  = (int*)carve((size_t)131072 * 4);
  unsigned short* A1   = (unsigned short*)carve((size_t)32 * 130 * 130 * 128 * 2);
  unsigned short* A2   = (unsigned short*)carve((size_t)32 * 66 * 66 * 64 * 2);
  unsigned short* Zf   = (unsigned short*)carve((size_t)131072 * 64 * 2);
  unsigned short* D1in = (unsigned short*)carve((size_t)32 * 130 * 130 * 64 * 2);
  unsigned short* D2in = (unsigned short*)carve((size_t)32 * 258 * 258 * 128 * 2);
  unsigned short* G2in = (unsigned short*)carve((size_t)32 * 258 * 258 * 64 * 2);

  // ---- zero-init (pads, dilation holes, accumulators) -- every call, deterministic
  auto zero = [&](void* p, long long bytes) {
    k_zero<<<4096, 256, 0, stream>>>((uint4*)p, bytes >> 4);
  };
  zero(A1,   (long long)32 * 130 * 130 * 128 * 2);
  zero(A2,   (long long)32 * 66 * 66 * 64 * 2);
  zero(D1in, (long long)32 * 130 * 130 * 64 * 2);
  zero(D2in, (long long)32 * 258 * 258 * 128 * 2);
  zero(G2in, (long long)32 * 258 * 258 * 64 * 2);
  k_zero<<<1, 256, 0, stream>>>((uint4*)Counts, (512 * 4 + 16) >> 4);

  // ---- weight prep
  k_pack_oihw<<<512, 256, 0, stream>>>(ew2, W2p, 64, 128, 4, 4);
  k_pack_oihw<<<144, 256, 0, stream>>>(ew3, W3p, 64, 64, 3, 3);
  k_pack_convT<<<288, 256, 0, stream>>>(dw1, DW1p, 64, 128, 3, 3);
  k_pack_convT<<<288, 256, 0, stream>>>(dw2, DW2p, 128, 64, 3, 3);
  k_pack_convT<<<3, 256, 0, stream>>>(dw3, DW3p, 64, 1, 3, 3);
  k_pack_emb<<<2, 256, 0, stream>>>(emb, EmbT, EmbN, 64, 512);

  dim3 cblk(32, 4);

  // ---- encoder
  k_conv1<<<2048, 256, 0, stream>>>(x, ew1, eb1, A1);
  // conv2: 4x4 s2, Cin=128 -> [32,64,64,64] into padded A2  (8192 M-tiles / 4)
  k_conv_wmma<4, 4, 2, 128><<<dim3(2048, 1), cblk, 0, stream>>>(
      A1, 130, 130, W2p, eb2, 1, 64, 64, A2, 66, 66, 1, 1, 64);
  // conv3: 3x3 s1, Cin=64 -> z flat [131072][64]
  k_conv_wmma<3, 3, 1, 64><<<dim3(2048, 1), cblk, 0, stream>>>(
      A2, 66, 66, W3p, eb3, 0, 64, 64, Zf, 64, 64, 0, 1, 64);

  // ---- vector quantizer
  k_vq_argmin<<<8192, 32, 0, stream>>>(Zf, EmbT, EmbN, Idx);
  k_vq_gather<<<512, 256, 0, stream>>>(Idx, EmbT, Zf, D1in, Counts, SSE);
  k_finalize<<<1, 512, 0, stream>>>(Counts, SSE, out, out + 2097153);

  // ---- decoder (transposed convs as stride-1 convs over dilated inputs)
  // convT1: -> [32,128,128,128], written dilated (so=2) into D2in (32768 M-tiles / 4; OC=128 -> grid.y=2)
  k_conv_wmma<3, 3, 1, 64><<<dim3(8192, 2), cblk, 0, stream>>>(
      D1in, 130, 130, DW1p, db1, 1, 128, 128, D2in, 258, 258, 1, 2, 128);
  // convT2: -> [32,64,256,256], written padded into G2in (131072 M-tiles / 4)
  k_conv_wmma<3, 3, 1, 128><<<dim3(32768, 1), cblk, 0, stream>>>(
      D2in, 258, 258, DW2p, db2, 1, 256, 256, G2in, 258, 258, 1, 1, 64);
  // convT3: -> x_recon fp32 at out[1..2097152]
  k_convT3<<<8192, 256, 0, stream>>>(G2in, DW3p, db3, out + 1);
}
